// GPT_40724879901136
// MI455X (gfx1250) — matrix operand order repaired and checked by reading, not compiled, not run on hardware
//
#include <hip/hip_runtime.h>
#include <math.h>

// ---------------- problem constants (from reference) ----------------
#define L_  6
#define B_  2
#define T_  1024
#define D_  768
#define H_  12
#define DK_ 64
#define V_  50257
#define FF_ (4 * D_)
#define ROWS_ (B_ * T_)          // 2048 token rows

typedef __attribute__((ext_vector_type(16))) _Float16 v16h;
typedef __attribute__((ext_vector_type(8)))  float    v8f;

// ---- CDNA5 async global->LDS copy (ASYNCcnt) with compile-safe fallback ----
#if __has_builtin(__builtin_amdgcn_global_load_async_to_lds_b128)
#define HAS_ASYNC_LDS 1
typedef int v4i_ __attribute__((ext_vector_type(4)));
typedef __attribute__((address_space(1))) v4i_ gv4i;   // global int4
typedef __attribute__((address_space(3))) v4i_ lv4i;   // LDS int4
#else
#define HAS_ASYNC_LDS 0
#endif

__device__ __forceinline__ void wait_async0() {
#if __has_builtin(__builtin_amdgcn_s_wait_asynccnt)
  __builtin_amdgcn_s_wait_asynccnt(0);
#else
  asm volatile("s_wait_asynccnt 0x0" ::: "memory");
#endif
}

// =====================================================================
// Embedding gather: x[b,t,:] = emb[token[b,t], :]
// =====================================================================
__global__ __launch_bounds__(256) void k_embed(const int* __restrict__ tok,
                                               const float* __restrict__ emb,
                                               float* __restrict__ x) {
  int idx = blockIdx.x * 256 + threadIdx.x;
  if (idx >= ROWS_ * D_) return;
  int row = idx / D_, d = idx - row * D_;
  x[idx] = emb[(size_t)tok[row] * D_ + d];
}

// =====================================================================
// LayerNorm (population variance, eps=1e-5): one block per row
// =====================================================================
__global__ __launch_bounds__(256) void k_layernorm(const float* __restrict__ x,
                                                   const float* __restrict__ w,
                                                   const float* __restrict__ b,
                                                   float* __restrict__ y) {
  int row = blockIdx.x, tid = threadIdx.x;
  const float* xr = x + (size_t)row * D_;
  float s = 0.f, ss = 0.f;
  for (int d = tid; d < D_; d += 256) { float v = xr[d]; s += v; ss += v * v; }
  __shared__ float rs[256], rq[256];
  rs[tid] = s; rq[tid] = ss; __syncthreads();
  for (int o = 128; o; o >>= 1) {
    if (tid < o) { rs[tid] += rs[tid + o]; rq[tid] += rq[tid + o]; }
    __syncthreads();
  }
  float mean = rs[0] * (1.0f / D_);
  float var  = rq[0] * (1.0f / D_) - mean * mean;
  float rstd = rsqrtf(var + 1e-5f);
  float* yr = y + (size_t)row * D_;
  for (int d = tid; d < D_; d += 256) yr[d] = (xr[d] - mean) * rstd * w[d] + b[d];
}

// =====================================================================
// RoPE on q and k in place. Pair (j, j+32) within each head's 64 dims.
// =====================================================================
__global__ __launch_bounds__(256) void k_rope(float* __restrict__ q,
                                              float* __restrict__ k) {
  int idx = blockIdx.x * 256 + threadIdx.x;
  if (idx >= B_ * T_ * H_ * (DK_ / 2)) return;
  int j = idx & 31;
  int h = (idx >> 5) % H_;
  int t = (idx / (32 * H_)) % T_;
  int b = idx / (32 * H_ * T_);
  // inv_freq[j] = 10000^(-2j/64)
  float inv = __expf(-9.210340371976184f * ((float)(2 * j) / (float)DK_));
  float ang = (float)t * inv;
  float sn, cs;
  sincosf(ang, &sn, &cs);
  size_t base = ((size_t)(b * T_ + t)) * D_ + h * DK_;
  float q1 = q[base + j], q2 = q[base + j + 32];
  q[base + j]      = q1 * cs - q2 * sn;
  q[base + j + 32] = q2 * cs + q1 * sn;
  float k1 = k[base + j], k2 = k[base + j + 32];
  k[base + j]      = k1 * cs - k2 * sn;
  k[base + j + 32] = k2 * cs + k1 * sn;
}

// =====================================================================
// WMMA GEMM: C[M,N] = act(A[M,K] @ B + bias) + residual
//   ACT: 0 = none, 1 = exact GELU
//   BTR: false -> B is [K,N] row-major; true -> B element (k,n) = Bm[n*K+k]
// Block: 256 threads = 8 waves; 128x64 C tile; wave grid 4(M) x 2(N);
// each wave: 32x32 C via four v_wmma_f32_16x16x32_f16 accumulators
// (each A/B fragment reused twice -> half the LDS reads per FLOP).
// A tile staged raw-f32 via GLOBAL_LOAD_ASYNC_TO_LDS_B128 when available.
// =====================================================================
template <int ACT, bool BTR>
__global__ __launch_bounds__(256) void k_gemm(const float* __restrict__ A,
                                              const float* __restrict__ Bm,
                                              const float* __restrict__ bias,
                                              const float* __restrict__ res,
                                              float* __restrict__ C,
                                              int M, int N, int K) {
  __shared__ float    Asf[128][32];  // [m][k], raw f32 (async-copied)
  __shared__ _Float16 Bs[64][32];    // transposed tile: [n][k], f16
  int tid = threadIdx.x;
  int m0 = blockIdx.y * 128, n0 = blockIdx.x * 64;
  int w = tid >> 5, lane = tid & 31;
  int laneHi = lane >> 4, ln = lane & 15;
  int wm = w >> 1, wn = w & 1;     // wave grid 4x2

  v8f acc[2][2]; v8f zero8 = {};
  acc[0][0] = zero8; acc[0][1] = zero8; acc[1][0] = zero8; acc[1][1] = zero8;

  for (int k0 = 0; k0 < K; k0 += 32) {
    // ---- stage A tile: 128 rows x 32 k f32 (16 floats/thread) ----
    {
      int r = tid >> 1, c = (tid & 1) * 16;
      const float* src = A + (size_t)(m0 + r) * K + k0 + c;
#if HAS_ASYNC_LDS
      #pragma unroll
      for (int i = 0; i < 16; i += 4)
        __builtin_amdgcn_global_load_async_to_lds_b128(
            (gv4i*)(src + i), (lv4i*)&Asf[r][c + i], 0, 0);
#else
      #pragma unroll
      for (int i = 0; i < 16; ++i) Asf[r][c + i] = src[i];
#endif
    }
    // ---- stage B tile transposed into Bs[n][k] (f32 -> f16) ----
    if (!BTR) {
      int kk = tid >> 3, c = (tid & 7) * 8;            // 32 k-rows x 8 n-groups
      const float* src = Bm + (size_t)(k0 + kk) * N + n0 + c;
      #pragma unroll
      for (int i = 0; i < 8; ++i) {
        int n = c + i;
        float vv = (n0 + n < N) ? src[i] : 0.0f;
        Bs[n][kk] = (_Float16)vv;
      }
    } else {
      int n = tid >> 2, c = (tid & 3) * 8;             // 64 n-rows x 4 k-groups
      const float* src = Bm + (size_t)(n0 + n) * K + k0 + c;
      bool ok = (n0 + n) < N;
      #pragma unroll
      for (int i = 0; i < 8; ++i)
        Bs[n][c + i] = ok ? (_Float16)src[i] : (_Float16)0.0f;
    }
#if HAS_ASYNC_LDS
    wait_async0();    // producer's async A copies landed in LDS
#endif
    __syncthreads();

    // ---- fragments: contiguous k, same order for A and B ----
    v16h a[2], bb[2];
    #pragma unroll
    for (int mt = 0; mt < 2; ++mt)
      #pragma unroll
      for (int h = 0; h < 16; ++h) {
        int koff = (laneHi << 4) + h;  // A: lanes lo K0-15, hi K16-31
        a[mt][h] = (_Float16)Asf[wm * 32 + mt * 16 + ln][koff];
      }
    #pragma unroll
    for (int nt = 0; nt < 2; ++nt)
      #pragma unroll
      for (int h = 0; h < 16; ++h)                              // B: lanes lo K0-15, hi K16-31
        bb[nt][h] = Bs[wn * 32 + nt * 16 + ln][(laneHi << 4) + h];

    #pragma unroll
    for (int mt = 0; mt < 2; ++mt)
      #pragma unroll
      for (int nt = 0; nt < 2; ++nt)
        acc[mt][nt] = __builtin_amdgcn_wmma_f32_16x16x32_f16(
            false, a[mt], false, bb[nt], (short)0, acc[mt][nt], false, false);
    __syncthreads();
  }

  // ---- epilogue: bias + activation + residual ----
  #pragma unroll
  for (int mt = 0; mt < 2; ++mt) {
    int rowb = m0 + wm * 32 + mt * 16 + laneHi * 8;
    #pragma unroll
    for (int r = 0; r < 8; ++r) {
      int row = rowb + r;
      #pragma unroll
      for (int nt = 0; nt < 2; ++nt) {
        int col = n0 + wn * 32 + nt * 16 + ln;
        if (col >= N) continue;
        float v = acc[mt][nt][r];
        if (bias) v += bias[col];
        if (ACT == 1) v = 0.5f * v * (1.0f + erff(v * 0.7071067811865475f));
        size_t oidx = (size_t)row * N + col;
        if (res) v += res[oidx];
        C[oidx] = v;
      }
    }
  }
}

// =====================================================================
// Flash attention, causal, scale = 1/8. Block = (b, h, 128 q rows),
// 8 waves; wave w owns q rows [w*16, w*16+16) fully -> softmax stats
// stay within a 16-lane half-wave (shuffle reductions only).
// =====================================================================
__global__ __launch_bounds__(256) void k_attn(const float* __restrict__ q,
                                              const float* __restrict__ kx,
                                              const float* __restrict__ vx,
                                              float* __restrict__ att) {
  int qt = blockIdx.x, h = blockIdx.y, b = blockIdx.z;
  int tid = threadIdx.x;
  int w = tid >> 5, lane = tid & 31, laneHi = lane >> 4, ln = lane & 15;
  int qrow0 = qt * 128;

  __shared__ _Float16 Qs[128][64];     // q tile (row-major)
  __shared__ _Float16 Ks[64][64];      // k tile row-major == B^T layout for Q.K^T
  __shared__ _Float16 VTs[64][64];     // v tile transposed: [dk][key]
  __shared__ _Float16 Ps[8][16][64];   // per-wave probability tile

  // stage Q (f32 -> f16)
  {
    int r = tid >> 1, c = (tid & 1) * 32;
    const float* src = q + ((size_t)(b * T_ + qrow0 + r)) * D_ + h * DK_ + c;
    #pragma unroll
    for (int i = 0; i < 32; ++i) Qs[r][c + i] = (_Float16)src[i];
  }

  float m_i[8], l_i[8];
  v8f accO[4]; v8f zero8 = {};
  #pragma unroll
  for (int r = 0; r < 8; ++r) { m_i[r] = -3.0e38f; l_i[r] = 0.0f; }
  #pragma unroll
  for (int dt = 0; dt < 4; ++dt) accO[dt] = zero8;

  const float scale = 0.125f;  // 1/sqrt(64)
  int jmax = qt * 2 + 1;       // causal: key tiles up to q block

  for (int j = 0; j <= jmax; ++j) {
    // stage K tile
    {
      int r = tid >> 2, c = (tid & 3) * 16;
      const float* src = kx + ((size_t)(b * T_ + j * 64 + r)) * D_ + h * DK_ + c;
      #pragma unroll
      for (int i = 0; i < 16; ++i) Ks[r][c + i] = (_Float16)src[i];
    }
    // stage V tile transposed
    {
      int key = tid >> 2, c = (tid & 3) * 16;
      const float* src = vx + ((size_t)(b * T_ + j * 64 + key)) * D_ + h * DK_ + c;
      #pragma unroll
      for (int i = 0; i < 16; ++i) VTs[c + i][key] = (_Float16)src[i];
    }
    __syncthreads();

    // ---- S = Q K^T (16x64 per wave, 4 accumulators, 2 k-steps) ----
    v8f accS[4];
    #pragma unroll
    for (int nt = 0; nt < 4; ++nt) accS[nt] = zero8;
    #pragma unroll
    for (int kk = 0; kk < 2; ++kk) {
      v16h a;
      #pragma unroll
      for (int hh = 0; hh < 16; ++hh) {
        int koff = (laneHi << 4) + hh;
        a[hh] = Qs[w * 16 + ln][kk * 32 + koff];
      }
      #pragma unroll
      for (int nt = 0; nt < 4; ++nt) {
        v16h bb;
        #pragma unroll
        for (int hh = 0; hh < 16; ++hh)
          bb[hh] = Ks[nt * 16 + ln][kk * 32 + (laneHi << 4) + hh];
        accS[nt] = __builtin_amdgcn_wmma_f32_16x16x32_f16(
            false, a, false, bb, (short)0, accS[nt], false, false);
      }
    }

    // ---- online softmax (per-row, within 16-lane half-wave) ----
    #pragma unroll
    for (int r = 0; r < 8; ++r) {
      int rowg = qrow0 + w * 16 + r + laneHi * 8;
      float rowmax = -3.0e38f;
      #pragma unroll
      for (int nt = 0; nt < 4; ++nt) {
        int colg = j * 64 + nt * 16 + ln;
        float s = accS[nt][r] * scale;
        if (colg > rowg) s = -3.0e38f;    // causal mask (finite: no NaN)
        accS[nt][r] = s;
        rowmax = fmaxf(rowmax, s);
      }
      #pragma unroll
      for (int off = 8; off >= 1; off >>= 1)
        rowmax = fmaxf(rowmax, __shfl_xor(rowmax, off, 16));
      float mnew  = fmaxf(m_i[r], rowmax);
      float alpha = __expf(m_i[r] - mnew);
      float rowsum = 0.0f;
      #pragma unroll
      for (int nt = 0; nt < 4; ++nt) {
        float p = __expf(accS[nt][r] - mnew);
        Ps[w][r + laneHi * 8][nt * 16 + ln] = (_Float16)p;
        rowsum += p;
      }
      #pragma unroll
      for (int off = 8; off >= 1; off >>= 1)
        rowsum += __shfl_xor(rowsum, off, 16);
      l_i[r] = l_i[r] * alpha + rowsum;
      m_i[r] = mnew;
      #pragma unroll
      for (int dt = 0; dt < 4; ++dt) accO[dt][r] = accO[dt][r] * alpha;
    }

    // ---- O += P @ V (LDS ops are in-order within the wave: no barrier) ----
    #pragma unroll
    for (int kk = 0; kk < 2; ++kk) {
      v16h a;
      #pragma unroll
      for (int hh = 0; hh < 16; ++hh) {
        int koff = (laneHi << 4) + hh;
        a[hh] = Ps[w][ln][kk * 32 + koff];
      }
      #pragma unroll
      for (int dt = 0; dt < 4; ++dt) {
        v16h bb;
        #pragma unroll
        for (int hh = 0; hh < 16; ++hh)
          bb[hh] = VTs[dt * 16 + ln][kk * 32 + (laneHi << 4) + hh];
        accO[dt] = __builtin_amdgcn_wmma_f32_16x16x32_f16(
            false, a, false, bb, (short)0, accO[dt], false, false);
      }
    }
    __syncthreads();  // before next key tile overwrites Ks/VTs
  }

  // ---- normalize and store att[b, t, h*64 + dk] ----
  #pragma unroll
  for (int r = 0; r < 8; ++r) {
    int row = qrow0 + w * 16 + r + laneHi * 8;
    float inv = 1.0f / l_i[r];
    #pragma unroll
    for (int dt = 0; dt < 4; ++dt)
      att[((size_t)(b * T_ + row)) * D_ + h * DK_ + dt * 16 + ln] = accO[dt][r] * inv;
  }
}

// =====================================================================
// Host-side orchestration
// =====================================================================
extern "C" void kernel_launch(void* const* d_in, const int* in_sizes, int n_in,
                              void* d_out, int out_size, void* d_ws, size_t ws_size,
                              hipStream_t stream) {
  const int*   tok   = (const int*)d_in[0];
  const float* emb   = (const float*)d_in[1];
  const float* gamma = (const float*)d_in[2];
  const float* beta  = (const float*)d_in[3];
  const float* Wq = (const float*)d_in[4];  const float* bq = (const float*)d_in[5];
  const float* Wk = (const float*)d_in[6];  const float* bk = (const float*)d_in[7];
  const float* Wv = (const float*)d_in[8];  const float* bv = (const float*)d_in[9];
  const float* Wo = (const float*)d_in[10]; const float* bo = (const float*)d_in[11];
  const float* ln2w = (const float*)d_in[12]; const float* ln2b = (const float*)d_in[13];
  const float* W1 = (const float*)d_in[14]; const float* b1 = (const float*)d_in[15];
  const float* W2 = (const float*)d_in[16]; const float* b2 = (const float*)d_in[17];
  const float* lnw = (const float*)d_in[18]; const float* lnb = (const float*)d_in[19];
  float* out = (float*)d_out;

  // workspace layout (floats): x, xn, q, k, v, att (each B*T*D) + h1 (B*T*4D)
  // total = 6*1.5M + 6.3M floats ~= 60 MB
  float* ws = (float*)d_ws;
  const size_t BTD = (size_t)ROWS_ * D_;
  float* x  = ws;
  float* xn = x  + BTD;
  float* qb = xn + BTD;
  float* kb = qb + BTD;
  float* vb = kb + BTD;
  float* ab = vb + BTD;
  float* h1 = ab + BTD;   // ROWS_ * FF_

  dim3 blk(256);
  k_embed<<<dim3((ROWS_ * D_ + 255) / 256), blk, 0, stream>>>(tok, emb, x);

  for (int l = 0; l < L_; ++l) {
    const size_t WDD = (size_t)D_ * D_;
    k_layernorm<<<dim3(ROWS_), blk, 0, stream>>>(x, gamma + l * D_, beta + l * D_, xn);
    k_gemm<0, false><<<dim3(D_ / 64, ROWS_ / 128), blk, 0, stream>>>(
        xn, Wq + l * WDD, bq + l * D_, nullptr, qb, ROWS_, D_, D_);
    k_gemm<0, false><<<dim3(D_ / 64, ROWS_ / 128), blk, 0, stream>>>(
        xn, Wk + l * WDD, bk + l * D_, nullptr, kb, ROWS_, D_, D_);
    k_gemm<0, false><<<dim3(D_ / 64, ROWS_ / 128), blk, 0, stream>>>(
        xn, Wv + l * WDD, bv + l * D_, nullptr, vb, ROWS_, D_, D_);
    k_rope<<<dim3((B_ * T_ * H_ * 32 + 255) / 256), blk, 0, stream>>>(qb, kb);
    k_attn<<<dim3(T_ / 128, H_, B_), blk, 0, stream>>>(qb, kb, vb, ab);
    k_gemm<0, false><<<dim3(D_ / 64, ROWS_ / 128), blk, 0, stream>>>(
        ab, Wo + l * WDD, bo + l * D_, x, x, ROWS_, D_, D_);     // + residual
    k_layernorm<<<dim3(ROWS_), blk, 0, stream>>>(x, ln2w + l * D_, ln2b + l * D_, xn);
    k_gemm<1, false><<<dim3(FF_ / 64, ROWS_ / 128), blk, 0, stream>>>(
        xn, W1 + (size_t)l * D_ * FF_, b1 + l * FF_, nullptr, h1, ROWS_, FF_, D_); // GELU
    k_gemm<0, false><<<dim3(D_ / 64, ROWS_ / 128), blk, 0, stream>>>(
        h1, W2 + (size_t)l * FF_ * D_, b2 + l * D_, x, x, ROWS_, D_, FF_); // + residual
  }

  k_layernorm<<<dim3(ROWS_), blk, 0, stream>>>(x, lnw, lnb, xn);
  // tied classifier: logits = xn @ emb^T  -> B with (k,n) = emb[n*K+k]
  k_gemm<0, true><<<dim3((V_ + 63) / 64, ROWS_ / 128), blk, 0, stream>>>(
      xn, emb, nullptr, nullptr, out, ROWS_, V_, D_);
}
